// SAGE_DIFFPOOL_38680475468404
// MI455X (gfx1250) — matrix operand order repaired and checked by reading, not compiled, not run on hardware
//
#include <hip/hip_runtime.h>
#include <hip/hip_bf16.h>
#include <math.h>

#define Bg 64
#define Nn 512
#define Hh 256
#define Kc 4
#define NV (Bg*Nn)
#define EPS_BN 1e-5f

typedef __attribute__((ext_vector_type(16))) __bf16 v16bf;
typedef __attribute__((ext_vector_type(8)))  __bf16 bf16x8;
typedef __attribute__((ext_vector_type(8)))  float  v8f;

#if __has_builtin(__builtin_amdgcn_global_load_async_to_lds_b32)
#define ASYNC_LDS 1
typedef __attribute__((address_space(1))) int g1int;
typedef __attribute__((address_space(3))) int l3int;
#else
#define ASYNC_LDS 0
#endif

// ---------------------------------------------------------------------------
// WMMA GEMM:  C[M,N] = A[M,K](bf16 row-major) @ B[K,N](bf16 row-major) + bias
// Block = 8 waves; each wave owns a 16x16 output tile in the same 16-col panel.
// Per 32-K step: the 32x16 B tile is cooperatively staged to LDS (async
// global->LDS when available), lanes gather B fragments from LDS; A fragments
// are two contiguous 16B vector loads per lane (ISA 7.12.2 layout).
// ---------------------------------------------------------------------------
__global__ __launch_bounds__(256)
void k_gemm_bf16(const __bf16* __restrict__ A, const __bf16* __restrict__ Bm,
                 const float* __restrict__ bias, float* __restrict__ C,
                 int M, int N, int K, size_t sA, size_t sB, size_t sC)
{
  __shared__ __bf16 Bt[32][18];            // +2 pad
  const int bz = blockIdx.z;
  A  += (size_t)bz * sA;
  Bm += (size_t)bz * sB;
  C  += (size_t)bz * sC;
  const int tid  = threadIdx.x;
  const int wave = tid >> 5, lane = tid & 31;
  const int r = lane & 15, hi = lane >> 4;
  const int bcol0 = blockIdx.x * 16;
  const int tilem = blockIdx.y * 8 + wave;
  const bool valid = (tilem * 16) < M;     // wave-uniform
  const __bf16* ap = A + (size_t)(tilem * 16 + r) * K;
  const int lkk = tid >> 3;                // 0..31: tile row loaded by this thread
  const int lc  = tid & 7;                 // 0..7 : dword within row
  v8f acc = {0.f,0.f,0.f,0.f,0.f,0.f,0.f,0.f};
  for (int k0 = 0; k0 < K; k0 += 32) {
    __syncthreads();
    const uint32_t* gp = (const uint32_t*)(Bm + (size_t)(k0 + lkk) * N + bcol0) + lc;
#if ASYNC_LDS
    __builtin_amdgcn_global_load_async_to_lds_b32(
        (g1int*)gp, (l3int*)&Bt[lkk][lc * 2], 0, 0);
#if __has_builtin(__builtin_amdgcn_s_wait_asynccnt)
    __builtin_amdgcn_s_wait_asynccnt(0);
#else
    asm volatile("s_wait_asynccnt 0x0" ::: "memory");
#endif
#else
    *(uint32_t*)&Bt[lkk][lc * 2] = *gp;
#endif
    __syncthreads();
    if (valid) {
      if (k0 + 32 < K) __builtin_prefetch((const void*)(ap + k0 + 32 + 8*hi), 0, 1);
      bf16x8 alo = *(const bf16x8*)(ap + k0 + 8*hi);        // K = 8hi .. 8hi+7
      bf16x8 ahi = *(const bf16x8*)(ap + k0 + 16 + 8*hi);   // K = 16+8hi .. 16+8hi+7
      v16bf a = __builtin_shufflevector(alo, ahi, 0,1,2,3,4,5,6,7,8,9,10,11,12,13,14,15);
      v16bf b;
#pragma unroll
      for (int e = 0; e < 16; ++e) b[e] = Bt[(e < 8) ? (8*hi + e) : (16 + 8*hi + (e - 8))][r];
      acc = __builtin_amdgcn_wmma_f32_16x16x32_bf16(false, a, false, b, (short)0, acc,
                                                    false, false);
    }
  }
  if (valid) {
    float badd = bias ? bias[bcol0 + r] : 0.f;
#pragma unroll
    for (int j = 0; j < 8; ++j)
      C[(size_t)(tilem*16 + j + 8*hi) * N + (bcol0 + r)] = acc[j] + badd;
  }
}

// ---- adj[B,N,N] f32 -> adjT[B,N,N] bf16 (transposed) via LDS tile ----------
__global__ void k_adjT(const float* __restrict__ adj, __bf16* __restrict__ adjT)
{
  __shared__ float t[16][17];
  const int b  = blockIdx.z;
  const int s0 = blockIdx.x * 16, d0 = blockIdx.y * 16;
  const int tx = threadIdx.x, ty = threadIdx.y;
  t[ty][tx] = adj[(size_t)b*Nn*Nn + (size_t)(s0+ty)*Nn + d0 + tx];
  __syncthreads();
  adjT[(size_t)b*Nn*Nn + (size_t)(d0+ty)*Nn + s0 + tx] = (__bf16)t[tx][ty];
}

// ---- deg[n] = 1 + in-degree(n) = 1 + row-sum of adjT ----------------------
__global__ void k_deg(const __bf16* __restrict__ adjT, float* __restrict__ deg)
{
  int n = blockIdx.x * blockDim.x + threadIdx.x;
  const __bf16* row = adjT + (size_t)n * Nn;
  float s = 1.0f;
  for (int i = 0; i < Nn; ++i) s += (float)row[i];
  deg[n] = s;
}

// ---- agg3 = (adjT @ x + x) / deg  (3-wide input features) ----------------
__global__ void k_aggx(const __bf16* __restrict__ adjT, const float* __restrict__ x,
                       const float* __restrict__ deg, float* __restrict__ agg3)
{
  int n = blockIdx.x * blockDim.x + threadIdx.x; // global node b*N+d
  int b = n >> 9;
  const __bf16* row = adjT + (size_t)n * Nn;
  const float* xb = x + (size_t)b * Nn * 3;
  float a0 = 0.f, a1 = 0.f, a2 = 0.f;
  for (int s = 0; s < Nn; ++s) {
    float w = (float)row[s];
    a0 += w * xb[s*3+0];
    a1 += w * xb[s*3+1];
    a2 += w * xb[s*3+2];
  }
  float inv = 1.0f / deg[n];
  agg3[n*3+0] = (a0 + x[n*3+0]) * inv;
  agg3[n*3+1] = (a1 + x[n*3+1]) * inv;
  agg3[n*3+2] = (a2 + x[n*3+2]) * inv;
}

// ---- out[NV,256] = agg3[NV,3] @ W[3,256] + b ------------------------------
__global__ void k_lin3(const float* __restrict__ agg3, const float* __restrict__ W,
                       const float* __restrict__ bias, float* __restrict__ out)
{
  int idx = blockIdx.x * blockDim.x + threadIdx.x;
  int n = idx >> 8, c = idx & 255;
  float v = bias[c] + agg3[n*3+0]*W[c] + agg3[n*3+1]*W[256+c] + agg3[n*3+2]*W[512+c];
  out[idx] = v;
}

// ---- BatchNorm stats: one block per channel -------------------------------
__global__ void k_bn_stats(const float* __restrict__ X, int rows, int C,
                           float* __restrict__ mean, float* __restrict__ var)
{
  __shared__ float s1[256], s2[256];
  int c = blockIdx.x;
  float a = 0.f, b = 0.f;
  for (int r = threadIdx.x; r < rows; r += blockDim.x) {
    float v = X[(size_t)r * C + c];
    a += v; b += v * v;
  }
  s1[threadIdx.x] = a; s2[threadIdx.x] = b;
  __syncthreads();
  for (int st = 128; st > 0; st >>= 1) {
    if (threadIdx.x < st) { s1[threadIdx.x] += s1[threadIdx.x+st]; s2[threadIdx.x] += s2[threadIdx.x+st]; }
    __syncthreads();
  }
  if (threadIdx.x == 0) {
    float m = s1[0] / rows;
    mean[c] = m;
    var[c]  = s2[0] / rows - m * m;
  }
}

__global__ void k_bn_apply(const float* __restrict__ X, const float* __restrict__ mean,
                           const float* __restrict__ var, const float* __restrict__ g,
                           const float* __restrict__ bb, int n, int C,
                           float* __restrict__ outF, __bf16* __restrict__ outB)
{
  int i = blockIdx.x * blockDim.x + threadIdx.x;
  if (i >= n) return;
  int c = i % C;
  float v = (X[i] - mean[c]) * rsqrtf(var[c] + EPS_BN) * g[c] + bb[c];
  if (outF) outF[i] = v;
  if (outB) outB[i] = (__bf16)v;
}

// ---- SAGE post: v = (agg + self)/deg -> f32 (in place) and bf16 -----------
__global__ void k_sage_post(float* __restrict__ agg, const float* __restrict__ selfa,
                            const float* __restrict__ deg, __bf16* __restrict__ outB,
                            int n, int C)
{
  int i = blockIdx.x * blockDim.x + threadIdx.x;
  if (i >= n) return;
  float v = (agg[i] + selfa[i]) / deg[i / C];
  agg[i] = v;
  outB[i] = (__bf16)v;
}

__global__ void k_f2b(const float* __restrict__ in, __bf16* __restrict__ out, int n)
{
  int i = blockIdx.x * blockDim.x + threadIdx.x;
  if (i < n) out[i] = (__bf16)in[i];
}

__global__ void k_relu_f2b(const float* __restrict__ in, __bf16* __restrict__ out, int n)
{
  int i = blockIdx.x * blockDim.x + threadIdx.x;
  if (i < n) out[i] = (__bf16)fmaxf(in[i], 0.f);
}

// ---- small dense linear (used for 256 -> 4 pooling projection) ------------
__global__ void k_lin_small(const float* __restrict__ X, const float* __restrict__ W,
                            const float* __restrict__ bias, float* __restrict__ out,
                            int rows, int Kd, int C)
{
  int idx = blockIdx.x * blockDim.x + threadIdx.x;
  if (idx >= rows * C) return;
  int row = idx / C, c = idx % C;
  float a = bias[c];
  const float* xr = X + (size_t)row * Kd;
  for (int k = 0; k < Kd; ++k) a += xr[k] * W[(size_t)k * C + c];
  out[idx] = a;
}

__global__ void k_softmax4(float* __restrict__ sm)
{
  int n = blockIdx.x * blockDim.x + threadIdx.x;
  float* p = sm + (size_t)n * 4;
  float m = fmaxf(fmaxf(p[0], p[1]), fmaxf(p[2], p[3]));
  float e0 = expf(p[0]-m), e1 = expf(p[1]-m), e2 = expf(p[2]-m), e3 = expf(p[3]-m);
  float inv = 1.0f / (e0+e1+e2+e3);
  p[0]=e0*inv; p[1]=e1*inv; p[2]=e2*inv; p[3]=e3*inv;
}

// ---- entropy loss partials (per block) ------------------------------------
__global__ void k_ent(const float* __restrict__ sm, float* __restrict__ pbuf)
{
  __shared__ float red[256];
  int i = blockIdx.x * blockDim.x + threadIdx.x;
  float s = 0.f;
  if (i < NV) {
    for (int k = 0; k < 4; ++k) {
      float p = sm[(size_t)i*4 + k];
      s += -p * logf(p + 1e-15f);
    }
  }
  red[threadIdx.x] = s; __syncthreads();
  for (int st = 128; st > 0; st >>= 1) {
    if (threadIdx.x < st) red[threadIdx.x] += red[threadIdx.x+st];
    __syncthreads();
  }
  if (threadIdx.x == 0) pbuf[blockIdx.x] = red[0];
}

// ---- link loss partials: sum (adj - sm@smT)^2 -----------------------------
__global__ void k_link(const float* __restrict__ adj, const float* __restrict__ sm,
                       float* __restrict__ pbuf)
{
  __shared__ float red[256];
  size_t idx = (size_t)blockIdx.x * blockDim.x + threadIdx.x;
  int b = (int)(idx / ((size_t)Nn*Nn));
  int rem = (int)(idx % ((size_t)Nn*Nn));
  int n = rem / Nn, m = rem % Nn;
  const float* sn = sm + (size_t)(b*Nn + n) * 4;
  const float* smm = sm + (size_t)(b*Nn + m) * 4;
  float dot = sn[0]*smm[0] + sn[1]*smm[1] + sn[2]*smm[2] + sn[3]*smm[3];
  float d = adj[idx] - dot;
  red[threadIdx.x] = d * d; __syncthreads();
  for (int st = 128; st > 0; st >>= 1) {
    if (threadIdx.x < st) red[threadIdx.x] += red[threadIdx.x+st];
    __syncthreads();
  }
  if (threadIdx.x == 0) pbuf[blockIdx.x] = red[0];
}

__global__ void k_reduce(const float* __restrict__ in, int n, float* __restrict__ out)
{
  __shared__ float s[256];
  float a = 0.f;
  for (int i = threadIdx.x; i < n; i += 256) a += in[i];
  s[threadIdx.x] = a; __syncthreads();
  for (int st = 128; st > 0; st >>= 1) {
    if (threadIdx.x < st) s[threadIdx.x] += s[threadIdx.x+st];
    __syncthreads();
  }
  if (threadIdx.x == 0) out[0] = s[0];
}

// ---- px[b,k,:] = sum_n sm[b,n,k] * h[b,n,:] -------------------------------
__global__ void k_px(const float* __restrict__ sm, const float* __restrict__ h,
                     float* __restrict__ px)
{
  int b = blockIdx.x, k = blockIdx.y, c = threadIdx.x;
  float acc = 0.f;
  for (int n = 0; n < Nn; ++n)
    acc += sm[(size_t)(b*Nn+n)*4 + k] * h[(size_t)(b*Nn+n)*Hh + c];
  px[((size_t)b*Kc + k)*Hh + c] = acc;
}

// ---- t[b,n,j] = sum_m adj[b,n,m] sm[b,m,j] --------------------------------
__global__ void k_adjsm(const float* __restrict__ adj, const float* __restrict__ sm,
                        float* __restrict__ t)
{
  int idx = blockIdx.x * blockDim.x + threadIdx.x; // NV*4
  int ng = idx >> 2, j = idx & 3;
  int b = ng >> 9;
  const float* arow = adj + (size_t)ng * Nn;
  float s = 0.f;
  for (int m = 0; m < Nn; ++m) s += arow[m] * sm[(size_t)(b*Nn + m)*4 + j];
  t[idx] = s;
}

// ---- padj[b,i,j] = sum_n sm[b,n,i] t[b,n,j] -------------------------------
__global__ void k_padj(const float* __restrict__ sm, const float* __restrict__ t,
                       float* __restrict__ padj)
{
  int b = blockIdx.x, tt = threadIdx.x; // 16 threads
  int i = tt >> 2, j = tt & 3;
  float s = 0.f;
  for (int n = 0; n < Nn; ++n)
    s += sm[(size_t)(b*Nn+n)*4 + i] * t[(size_t)(b*Nn+n)*4 + j];
  padj[b*16 + tt] = s;
}

// ---- dense SAGE on pooled graph (K=4 nodes, H=256) ------------------------
__global__ void k_sage_dense(const float* __restrict__ Xin, const float* __restrict__ padj,
                             const float* __restrict__ W, const float* __restrict__ bias,
                             float* __restrict__ out)
{
  __shared__ float lx[Kc][Hh];
  __shared__ float mrow[Kc][Kc];
  __shared__ float minv[Kc];
  int b = blockIdx.x, t = threadIdx.x;
  for (int i = 0; i < Kc; ++i) lx[i][t] = Xin[((size_t)b*Kc + i)*Hh + t];
  if (t < 16) {
    int i = t >> 2, j = t & 3;
    float a = padj[b*16 + t];
    mrow[i][j] = (i == j) ? 1.f : (a != 0.f ? 1.f : 0.f);
  }
  __syncthreads();
  if (t < Kc) {
    float s = 0.f;
    for (int j = 0; j < Kc; ++j) s += mrow[t][j];
    minv[t] = 1.f / s;
  }
  __syncthreads();
  float acc[Kc];
  for (int i = 0; i < Kc; ++i) acc[i] = bias[t];
  for (int k = 0; k < Hh; ++k) {
    float w = W[(size_t)k*Hh + t];
#pragma unroll
    for (int i = 0; i < Kc; ++i) {
      float aggik = (mrow[i][0]*lx[0][k] + mrow[i][1]*lx[1][k] +
                     mrow[i][2]*lx[2][k] + mrow[i][3]*lx[3][k]) * minv[i];
      acc[i] += aggik * w;
    }
  }
  for (int i = 0; i < Kc; ++i) out[((size_t)b*Kc + i)*Hh + t] = acc[i];
}

// ---- final fc3: out[64,6] = relu(X)@W + b ---------------------------------
__global__ void k_fc3(const float* __restrict__ X, const float* __restrict__ W,
                      const float* __restrict__ bias, float* __restrict__ out)
{
  int i = blockIdx.x * blockDim.x + threadIdx.x;
  if (i >= 64*6) return;
  int r = i / 6, c = i % 6;
  float a = bias[c];
  for (int k = 0; k < 128; ++k) a += fmaxf(X[r*128 + k], 0.f) * W[k*6 + c];
  out[i] = a;
}

__global__ void k_fin(const float* __restrict__ loss, float* __restrict__ out)
{
  if (threadIdx.x == 0) {
    float link = sqrtf(loss[0]) / (float)((size_t)Bg * Nn * Nn);
    float ent  = loss[1] / (float)NV;
    out[0] = (link * 1000.f + ent) * 100.f;
  }
}

// ===========================================================================
extern "C" void kernel_launch(void* const* d_in, const int* in_sizes, int n_in,
                              void* d_out, int out_size, void* d_ws, size_t ws_size,
                              hipStream_t stream)
{
  (void)in_sizes; (void)n_in; (void)out_size; (void)ws_size;
  const float* x   = (const float*)d_in[0];
  const float* adj = (const float*)d_in[2];
  const float* conv11_w = (const float*)d_in[3];
  const float* conv11_b = (const float*)d_in[4];
  const float* conv12_w = (const float*)d_in[5];
  const float* conv12_b = (const float*)d_in[6];
  const float* conv13_w = (const float*)d_in[7];
  const float* conv13_b = (const float*)d_in[8];
  const float* pconv11_w = (const float*)d_in[9];
  const float* pconv11_b = (const float*)d_in[10];
  const float* pconv12_w = (const float*)d_in[11];
  const float* pconv12_b = (const float*)d_in[12];
  const float* conv21_w = (const float*)d_in[13];
  const float* conv21_b = (const float*)d_in[14];
  const float* conv22_w = (const float*)d_in[15];
  const float* conv22_b = (const float*)d_in[16];
  const float* fc1_w = (const float*)d_in[17];
  const float* fc1_b = (const float*)d_in[18];
  const float* fc2_w = (const float*)d_in[19];
  const float* fc2_b = (const float*)d_in[20];
  const float* fc3_w = (const float*)d_in[21];
  const float* fc3_b = (const float*)d_in[22];
  const float* n11_g = (const float*)d_in[23];  const float* n11_b = (const float*)d_in[24];
  const float* n12_g = (const float*)d_in[25];  const float* n12_b = (const float*)d_in[26];
  const float* n13_g = (const float*)d_in[27];  const float* n13_b = (const float*)d_in[28];
  const float* np11_g = (const float*)d_in[29]; const float* np11_b = (const float*)d_in[30];
  const float* np12_g = (const float*)d_in[31]; const float* np12_b = (const float*)d_in[32];
  const float* n21_g = (const float*)d_in[33];  const float* n21_b = (const float*)d_in[34];
  const float* n22_g = (const float*)d_in[35];  const float* n22_b = (const float*)d_in[36];
  const float* bn1_g = (const float*)d_in[37];  const float* bn1_b = (const float*)d_in[38];
  const float* bn2_g = (const float*)d_in[39];  const float* bn2_b = (const float*)d_in[40];
  float* out = (float*)d_out;

  // ---- workspace carve ----
  char* wp = (char*)d_ws;
  auto carve = [&](size_t bytes) -> void* {
    void* q = (void*)wp;
    wp += (bytes + 255) & ~(size_t)255;
    return q;
  };
  __bf16* adjT = (__bf16*)carve((size_t)Bg*Nn*Nn*2);
  __bf16* Abf  = (__bf16*)carve((size_t)NV*Hh*2);
  float*  F0   = (float*) carve((size_t)NV*Hh*4);
  float*  F1   = (float*) carve((size_t)NV*Hh*4);
  float*  F2   = (float*) carve((size_t)NV*Hh*4);
  __bf16* wbf  = (__bf16*)carve((size_t)1024*256*2);
  float*  deg  = (float*) carve((size_t)NV*4);
  float*  agg3 = (float*) carve((size_t)NV*3*4);
  float*  sm   = (float*) carve((size_t)NV*4*4);
  float*  tb   = (float*) carve((size_t)NV*4*4);
  float*  px   = (float*) carve((size_t)Bg*Kc*Hh*4);
  float*  padj = (float*) carve((size_t)Bg*16*4);
  float*  h2a  = (float*) carve((size_t)Bg*Kc*Hh*4);
  float*  h2b  = (float*) carve((size_t)Bg*Kc*Hh*4);
  __bf16* cobf = (__bf16*)carve((size_t)Bg*Kc*Hh*2);
  float*  o1   = (float*) carve((size_t)64*256*4);
  float*  o1b  = (float*) carve((size_t)64*256*4);
  __bf16* o1bf = (__bf16*)carve((size_t)64*256*2);
  float*  o2   = (float*) carve((size_t)64*128*4);
  float*  o2b  = (float*) carve((size_t)64*128*4);
  float*  mean = (float*) carve(256*4);
  float*  var  = (float*) carve(256*4);
  float*  loss = (float*) carve(2*4);
  float*  pbuf = (float*) carve((size_t)65536*4);

  auto bn = [&](const float* X, int rows, int C, const float* g, const float* b,
                float* outF, __bf16* outB) {
    k_bn_stats<<<C, 256, 0, stream>>>(X, rows, C, mean, var);
    int n = rows * C;
    k_bn_apply<<<(n+255)/256, 256, 0, stream>>>(X, mean, var, g, b, n, C, outF, outB);
  };
  const size_t sAdj = (size_t)Nn*Nn, sAct = (size_t)Nn*Hh;

  // ---- setup: adjT, deg, shared input aggregation ----
  k_adjT<<<dim3(32,32,Bg), dim3(16,16), 0, stream>>>(adj, adjT);
  k_deg <<<NV/256, 256, 0, stream>>>(adjT, deg);
  k_aggx<<<NV/256, 256, 0, stream>>>(adjT, x, deg, agg3);

  // ---- conv11 (3->256) + norm11 ----
  k_lin3<<<NV, 256, 0, stream>>>(agg3, conv11_w, conv11_b, F0);
  bn(F0, NV, Hh, n11_g, n11_b, F1, Abf);

  // ---- conv12 (256->256) + norm12 ----
  k_gemm_bf16<<<dim3(Hh/16, Nn/128, Bg), 256, 0, stream>>>(adjT, Abf, nullptr, F0, Nn, Hh, Nn, sAdj, sAct, sAct);
  k_sage_post<<<NV*Hh/256, 256, 0, stream>>>(F0, F1, deg, Abf, NV*Hh, Hh);
  k_f2b<<<(Hh*Hh)/256, 256, 0, stream>>>(conv12_w, wbf, Hh*Hh);
  k_gemm_bf16<<<dim3(Hh/16, NV/128, 1), 256, 0, stream>>>(Abf, wbf, conv12_b, F0, NV, Hh, Hh, 0, 0, 0);
  bn(F0, NV, Hh, n12_g, n12_b, F1, Abf);

  // ---- conv13 (256->256) + norm13 -> h final in F2 ----
  k_gemm_bf16<<<dim3(Hh/16, Nn/128, Bg), 256, 0, stream>>>(adjT, Abf, nullptr, F0, Nn, Hh, Nn, sAdj, sAct, sAct);
  k_sage_post<<<NV*Hh/256, 256, 0, stream>>>(F0, F1, deg, Abf, NV*Hh, Hh);
  k_f2b<<<(Hh*Hh)/256, 256, 0, stream>>>(conv13_w, wbf, Hh*Hh);
  k_gemm_bf16<<<dim3(Hh/16, NV/128, 1), 256, 0, stream>>>(Abf, wbf, conv13_b, F0, NV, Hh, Hh, 0, 0, 0);
  bn(F0, NV, Hh, n13_g, n13_b, F2, (__bf16*)nullptr);

  // ---- pooling path: pool_conv11 (3->256) + norm_p11 ----
  k_lin3<<<NV, 256, 0, stream>>>(agg3, pconv11_w, pconv11_b, F0);
  bn(F0, NV, Hh, np11_g, np11_b, F1, Abf);

  // ---- pool_conv12 (256->4) + norm_p12 + softmax ----
  k_gemm_bf16<<<dim3(Hh/16, Nn/128, Bg), 256, 0, stream>>>(adjT, Abf, nullptr, F0, Nn, Hh, Nn, sAdj, sAct, sAct);
  k_sage_post<<<NV*Hh/256, 256, 0, stream>>>(F0, F1, deg, Abf, NV*Hh, Hh);
  k_lin_small<<<(NV*4)/256, 256, 0, stream>>>(F0, pconv12_w, pconv12_b, sm, NV, Hh, 4);
  bn(sm, NV, 4, np12_g, np12_b, sm, (__bf16*)nullptr);
  k_softmax4<<<NV/256, 256, 0, stream>>>(sm);

  // ---- losses (deterministic two-stage reductions) ----
  k_link<<<(Bg*Nn*Nn)/256, 256, 0, stream>>>(adj, sm, pbuf);
  k_reduce<<<1, 256, 0, stream>>>(pbuf, (Bg*Nn*Nn)/256, loss + 0);
  k_ent<<<NV/256, 256, 0, stream>>>(sm, pbuf);
  k_reduce<<<1, 256, 0, stream>>>(pbuf, NV/256, loss + 1);

  // ---- diff-pool: px, padj ----
  k_px<<<dim3(Bg, Kc), 256, 0, stream>>>(sm, F2, px);
  k_adjsm<<<(NV*4)/256, 256, 0, stream>>>(adj, sm, tb);
  k_padj<<<Bg, 16, 0, stream>>>(sm, tb, padj);

  // ---- pooled convs (dense SAGE) + norms ----
  k_sage_dense<<<Bg, 256, 0, stream>>>(px, padj, conv21_w, conv21_b, h2a);
  bn(h2a, Bg*Kc, Hh, n21_g, n21_b, h2b, (__bf16*)nullptr);
  k_sage_dense<<<Bg, 256, 0, stream>>>(h2b, padj, conv22_w, conv22_b, h2a);
  bn(h2a, Bg*Kc, Hh, n22_g, n22_b, h2b, (__bf16*)nullptr);   // h2b == co [64,1024]

  // ---- fc1 [64,1024]@[1024,256] + bn1 ----
  k_relu_f2b<<<(Bg*Kc*Hh)/256, 256, 0, stream>>>(h2b, cobf, Bg*Kc*Hh);
  k_f2b<<<(1024*256)/256, 256, 0, stream>>>(fc1_w, wbf, 1024*256);
  k_gemm_bf16<<<dim3(256/16, 1, 1), 256, 0, stream>>>(cobf, wbf, fc1_b, o1, 64, 256, 1024, 0, 0, 0);
  bn(o1, 64, 256, bn1_g, bn1_b, o1b, (__bf16*)nullptr);

  // ---- fc2 [64,256]@[256,128] + bn2 ----
  k_relu_f2b<<<(64*256)/256, 256, 0, stream>>>(o1b, o1bf, 64*256);
  k_f2b<<<(256*128)/256, 256, 0, stream>>>(fc2_w, wbf, 256*128);
  k_gemm_bf16<<<dim3(128/16, 1, 1), 256, 0, stream>>>(o1bf, wbf, fc2_b, o2, 64, 128, 256, 0, 0, 0);
  bn(o2, 64, 128, bn2_g, bn2_b, o2b, (__bf16*)nullptr);

  // ---- fc3 + regularizer ----
  k_fc3<<<2, 192, 0, stream>>>(o2b, fc3_w, fc3_b, out);
  k_fin<<<1, 32, 0, stream>>>(loss, out + 384);
}
